// JModule_67972152426945
// MI455X (gfx1250) — matrix operand order repaired and checked by reading, not compiled, not run on hardware
//
#include <hip/hip_runtime.h>
#include <hip/hip_bf16.h>

typedef __attribute__((ext_vector_type(16))) __bf16 v16bf;
typedef __attribute__((ext_vector_type(8)))  float  v8f;
typedef __attribute__((ext_vector_type(4)))  __bf16 v4bf;

#define Bz 32
#define Sz 1024
#define Hz 512
#define NWG 16
#define NTHREADS 256

// ---- workspace layout (bytes) ----
#define CTR_OFF   0
#define HBUF_OFF  4096                                   // 2 x (32x512) bf16 = 64 KB
#define WPK_OFF   (HBUF_OFF + 2*Bz*Hz*2)                 // packed weights: 2 x 16 x 128 x 1024 bf16 = 8 MB
#define ACT0_OFF  (WPK_OFF + (size_t)2*16*128*1024*2)    // x as bf16, (S,32,512) = 32 MB
#define ACT1_OFF  (ACT0_OFF + (size_t)Sz*Bz*Hz*2)        // layer0 output bf16, (S,32,512) = 32 MB

// ---- LDS layout (bytes) ----
#define LDS_W_BYTES (128*1024*2)               // 256 KB weight slice (column-major: [n][k])
#define LDS_Z_OFF   LDS_W_BYTES                // 16 tiles * 256 f32 = 16 KB
#define LDS_C_OFF   (LDS_W_BYTES + 16*1024)    // c state 32x32 f32 = 4 KB
#define LDS_BYTES   (LDS_C_OFF + 4*1024)       // 282624 B <= 320 KB per WGP

// A-fragment loader (16-bit A 16x32 per ISA layout): two contiguous 8-elem runs,
// run0 at k0, run1 at k0+16 (k0 already includes the lane>=16 half-offset of +8).
__device__ inline v16bf load_a_bf16(const __bf16* row, int k0) {
    union { uint4 q[2]; v16bf v; } u;
    u.q[0] = *(const uint4*)(row + k0);
    u.q[1] = *(const uint4*)(row + k0 + 16);
    return u.v;
}
// B-fragment loader (16-bit B 32x16): two 8-K runs at p and p+16 (column-major slice in LDS).
__device__ inline v16bf load_b_contig(const __bf16* p) {
    union { uint4 q[2]; v16bf v; } u;
    u.q[0] = *(const uint4*)(p);
    u.q[1] = *(const uint4*)(p + 16);
    return u.v;
}
// Read element (m, nl) of the 32x128 z block stored as 16 C/D-layout tiles.
__device__ inline float zread(const float* zbuf, int m, int nl) {
    int mt = m >> 4, mm = m & 15;
    int nt = nl >> 4, nn = nl & 15;
    int ln = nn + ((mm & 8) ? 16 : 0);
    return zbuf[((mt * 8 + nt) << 8) + ln * 8 + (mm & 7)];
}
__device__ inline float sigmoidf(float x) { return 1.0f / (1.0f + __expf(-x)); }

// Pack W = [Wx ; Wh] into per-workgroup column-major bf16 slices:
// wpk[l][wg][nl][k], nl = g*32 + hc, column h = wg*32 + hc, k in [0,1024).
__global__ void pack_weights(const float* __restrict__ Wx, const float* __restrict__ Wh,
                             __bf16* __restrict__ wpk) {
    int idx = blockIdx.x * blockDim.x + threadIdx.x;
    if (idx >= 2 * 16 * 128 * 1024) return;
    int k  = idx & 1023;
    int nl = (idx >> 10) & 127;
    int wg = (idx >> 17) & 15;
    int l  = idx >> 21;
    int g  = nl >> 5;
    int h  = wg * 32 + (nl & 31);
    float v = (k < Hz) ? Wx[(((size_t)l * 4 + g) * Hz + k) * Hz + h]
                       : Wh[(((size_t)l * 4 + g) * Hz + (k - Hz)) * Hz + h];
    wpk[idx] = (__bf16)v;
}

// One-time transpose+downconvert of x: (B,S,H) f32 -> (S,B,H) bf16.
// Pure bandwidth (~96 MB total), keeps all cvt out of the serial recurrence.
__global__ void pack_x(const float* __restrict__ x, __bf16* __restrict__ act0) {
    size_t idx = (size_t)blockIdx.x * blockDim.x + threadIdx.x;
    if (idx >= (size_t)Sz * Bz * Hz) return;
    int k  = (int)(idx & (Hz - 1));
    int bm = (int)((idx >> 9) & 31);
    int s  = (int)(idx >> 14);
    act0[idx] = (__bf16)x[((size_t)bm * Sz + s) * Hz + k];
}

__global__ void init_state(unsigned* ctr, __bf16* hbuf) {
    if (blockIdx.x == 0 && threadIdx.x == 0) *ctr = 0u;
    int n = 2 * Bz * Hz;
    for (int i = blockIdx.x * blockDim.x + threadIdx.x; i < n; i += gridDim.x * blockDim.x)
        hbuf[i] = (__bf16)0.0f;
}

// Persistent recurrence kernel: 16 WGs, each owns h-columns [wg*32, wg*32+32) (all 4 gates,
// so cell state c stays WG-private). Per step: z = [x_t ; h] @ Wslice (bf16 WMMA, f32 acc)
// -> gates -> c/h update -> 16-WG barrier. Weight slice stays LDS-resident for the layer.
__global__ __launch_bounds__(NTHREADS)
void lstm_layer(const __bf16* __restrict__ in_, const __bf16* __restrict__ wpk,
                const float* __restrict__ bias, __bf16* hbuf, unsigned* ctr,
                __bf16* __restrict__ ybf, float* __restrict__ yf32,
                float* __restrict__ h_last, float* __restrict__ c_last) {
    extern __shared__ char smem[];
    __bf16* Wlds = (__bf16*)smem;
    float*  zbuf = (float*)(smem + LDS_Z_OFF);
    float*  cbuf = (float*)(smem + LDS_C_OFF);

    const int tid  = threadIdx.x;
    const int wg   = blockIdx.x;   // 0..15
    const int lane = tid & 31;
    const int wv   = tid >> 5;     // wave id == N-tile id (8 tiles of 16 cols)

    // One-time: stage this WG's 256 KB weight slice into LDS with CDNA5 async
    // global->LDS B128 transfers (ASYNCcnt), then wait + barrier.
    {
        typedef __attribute__((address_space(3))) char lds_char_t;
        unsigned lds_base = (unsigned)(size_t)(lds_char_t*)smem;
        unsigned long long gbase = (unsigned long long)(const void*)(wpk + (size_t)wg * 128 * 1024);
#pragma unroll 1
        for (int it = 0; it < (LDS_W_BYTES / 16) / NTHREADS; ++it) {
            unsigned byteoff = (unsigned)((it * NTHREADS + tid) * 16);
            unsigned ldsaddr = lds_base + byteoff;
            asm volatile("global_load_async_to_lds_b128 %0, %1, %2"
                         :: "v"(ldsaddr), "v"(byteoff), "s"(gbase)
                         : "memory");
        }
        asm volatile("s_wait_asynccnt 0" ::: "memory");
    }
    for (int i = tid; i < Bz * 32; i += NTHREADS) cbuf[i] = 0.0f;

    // Elementwise-phase identity + hoisted biases (f,i,c,o for 4 columns each).
    const int m  = tid & 31;
    const int hq = tid >> 5;
    float bg[4][4];
#pragma unroll
    for (int g = 0; g < 4; ++g)
#pragma unroll
        for (int j = 0; j < 4; ++j)
            bg[g][j] = bias[g * Hz + wg * 32 + hq * 4 + j];

    __syncthreads();

    const int ml  = lane & 15;
    const int k0h = (lane & 16) ? 8 : 0;                 // A-fragment lane half-offset
    const __bf16* bcol = Wlds + (wv * 16 + ml) * 1024 + ((lane & 16) ? 8 : 0);

    for (int s = 0; s < Sz; ++s) {
        const __bf16* hcur  = hbuf + (s & 1) * (Bz * Hz);
        const __bf16* hrow0 = hcur + ml * Hz;
        const __bf16* hrow1 = hcur + (16 + ml) * Hz;
        const __bf16* row0  = in_ + ((size_t)s * Bz + ml) * Hz;        // act[s][b][:]
        const __bf16* row1  = in_ + ((size_t)s * Bz + 16 + ml) * Hz;
        if (s + 1 < Sz) {                                              // global_prefetch_b8
            __builtin_prefetch(row0 + Bz * Hz, 0, 3);
            __builtin_prefetch(row1 + Bz * Hz, 0, 3);
        }

        v8f acc0 = {0.f, 0.f, 0.f, 0.f, 0.f, 0.f, 0.f, 0.f};
        v8f acc1 = {0.f, 0.f, 0.f, 0.f, 0.f, 0.f, 0.f, 0.f};

        // K = 0..511: x_t contribution.
#pragma unroll 4
        for (int kt = 0; kt < 16; ++kt) {
            int k0 = kt * 32 + k0h;
            v16bf a0 = load_a_bf16(row0, k0);
            v16bf a1 = load_a_bf16(row1, k0);
            v16bf bb = load_b_contig(bcol + kt * 32);
            acc0 = __builtin_amdgcn_wmma_f32_16x16x32_bf16(false, a0, false, bb, (short)0, acc0, false, false);
            acc1 = __builtin_amdgcn_wmma_f32_16x16x32_bf16(false, a1, false, bb, (short)0, acc1, false, false);
        }
        // K = 512..1023: recurrent h contribution.
#pragma unroll 4
        for (int kt = 0; kt < 16; ++kt) {
            int k0 = kt * 32 + k0h;
            v16bf a0 = load_a_bf16(hrow0, k0);
            v16bf a1 = load_a_bf16(hrow1, k0);
            v16bf bb = load_b_contig(bcol + (16 + kt) * 32);
            acc0 = __builtin_amdgcn_wmma_f32_16x16x32_bf16(false, a0, false, bb, (short)0, acc0, false, false);
            acc1 = __builtin_amdgcn_wmma_f32_16x16x32_bf16(false, a1, false, bb, (short)0, acc1, false, false);
        }

        // Spill z tiles (C/D layout: one v8f per lane per tile).
        *(v8f*)(zbuf + (0 * 8 + wv) * 256 + lane * 8) = acc0;
        *(v8f*)(zbuf + (8 + wv) * 256 + lane * 8)     = acc1;
        __syncthreads();

        // Gates + cell update for columns hq*4..hq*4+3 of row m.
        float hv[4], cv[4];
#pragma unroll
        for (int j = 0; j < 4; ++j) {
            int col = hq * 4 + j;
            float zf = zread(zbuf, m, col)       + bg[0][j];
            float zi = zread(zbuf, m, 32 + col)  + bg[1][j];
            float zc = zread(zbuf, m, 64 + col)  + bg[2][j];
            float zo = zread(zbuf, m, 96 + col)  + bg[3][j];
            float gf = sigmoidf(zf);
            float gi = sigmoidf(zi);
            float gc = tanhf(zc);
            float go = sigmoidf(zo);
            float cn = gf * cbuf[col * 32 + m] + gi * gc;
            cbuf[col * 32 + m] = cn;
            cv[j] = cn;
            hv[j] = go * tanhf(cn);
        }

        v4bf h4;
#pragma unroll
        for (int j = 0; j < 4; ++j) h4[j] = (__bf16)hv[j];
        *(v4bf*)(hbuf + ((s + 1) & 1) * (Bz * Hz) + m * Hz + wg * 32 + hq * 4) = h4;
        if (ybf)
            *(v4bf*)(ybf + ((size_t)s * Bz + m) * Hz + wg * 32 + hq * 4) = h4;
        if (yf32) {
            float4 f4 = {hv[0], hv[1], hv[2], hv[3]};
            *(float4*)(yf32 + ((size_t)m * Sz + s) * Hz + wg * 32 + hq * 4) = f4;
        }
        if (s == Sz - 1) {
            float4 hf = {hv[0], hv[1], hv[2], hv[3]};
            float4 cf = {cv[0], cv[1], cv[2], cv[3]};
            *(float4*)(h_last + m * Hz + wg * 32 + hq * 4) = hf;
            *(float4*)(c_last + m * Hz + wg * 32 + hq * 4) = cf;
        }

        // Grid barrier across the 16 persistent WGs (release h writes, acquire for next read).
        __builtin_amdgcn_fence(__ATOMIC_RELEASE, "agent");
        __syncthreads();
        if (tid == 0) {
            __hip_atomic_fetch_add(ctr, 1u, __ATOMIC_RELEASE, __HIP_MEMORY_SCOPE_AGENT);
            unsigned tgt = (unsigned)NWG * (unsigned)(s + 1);
            while (__hip_atomic_load(ctr, __ATOMIC_ACQUIRE, __HIP_MEMORY_SCOPE_AGENT) < tgt)
                __builtin_amdgcn_s_sleep(2);
        }
        __syncthreads();
        __builtin_amdgcn_fence(__ATOMIC_ACQUIRE, "agent");
    }
}

extern "C" void kernel_launch(void* const* d_in, const int* in_sizes, int n_in,
                              void* d_out, int out_size, void* d_ws, size_t ws_size,
                              hipStream_t stream) {
    (void)in_sizes; (void)n_in; (void)out_size; (void)ws_size;
    const float* x  = (const float*)d_in[0];
    const float* Wx = (const float*)d_in[1];
    const float* Wh = (const float*)d_in[2];
    const float* b  = (const float*)d_in[3];
    float* out = (float*)d_out;

    char* ws = (char*)d_ws;
    unsigned* ctr = (unsigned*)(ws + CTR_OFF);
    __bf16* hbuf  = (__bf16*)(ws + HBUF_OFF);
    __bf16* wpk   = (__bf16*)(ws + WPK_OFF);
    __bf16* act0  = (__bf16*)(ws + ACT0_OFF);
    __bf16* act1  = (__bf16*)(ws + ACT1_OFF);

    float* y_out  = out;                                  // (B,S,H)
    float* h_last = out + (size_t)Bz * Sz * Hz;           // (L,B,H)
    float* c_last = h_last + (size_t)2 * Bz * Hz;         // (L,B,H)

    pack_weights<<<(2 * 16 * 128 * 1024) / NTHREADS, NTHREADS, 0, stream>>>(Wx, Wh, wpk);
    pack_x<<<(int)(((size_t)Sz * Bz * Hz) / NTHREADS), NTHREADS, 0, stream>>>(x, act0);

    // Layer 0: bf16 x input, bf16 activation output to scratch.
    init_state<<<64, NTHREADS, 0, stream>>>(ctr, hbuf);
    lstm_layer<<<NWG, NTHREADS, LDS_BYTES, stream>>>(
        act0, wpk, b, hbuf, ctr, act1, nullptr, h_last, c_last);

    // Layer 1: bf16 activation input, f32 output straight to d_out.
    init_state<<<64, NTHREADS, 0, stream>>>(ctr, hbuf);
    lstm_layer<<<NWG, NTHREADS, LDS_BYTES, stream>>>(
        act1, wpk + (size_t)16 * 128 * 1024, b + 4 * Hz, hbuf, ctr,
        nullptr, y_out, h_last + Bz * Hz, c_last + Bz * Hz);
}
